// SelfAttentionMasked_40046275068497
// MI455X (gfx1250) — matrix operand order repaired and checked by reading, not compiled, not run on hardware
//
#include <hip/hip_runtime.h>
#include <stdint.h>

// Problem dims (fixed by reference)
#define B_   4
#define N_   2048
#define DIM_ 1024
#define H_   16
#define DH_  64

typedef __attribute__((ext_vector_type(16))) __bf16 bf16x16;
typedef __attribute__((ext_vector_type(8)))  float  f32x8;

union Frag {
  uint32_t u[8];
  bf16x16  v;
};

__device__ __forceinline__ unsigned short f32_to_bf16(float f) {
  uint32_t u = __float_as_uint(f);
  u += 0x7FFFu + ((u >> 16) & 1u);   // round-to-nearest-even
  return (unsigned short)(u >> 16);
}

// A-matrix 16x32 bf16 K-offset per VGPR j (ISA 7.12.2)
__device__ __forceinline__ int koffA(int j, int hf) {
  return ((j >> 2) << 4) + hf * 8 + ((j & 3) << 1);
}
// B-matrix 32x16 bf16 K-offset per VGPR j: same K order as A
__device__ __forceinline__ int koffB(int j, int hf) {
  return ((j >> 2) << 4) + hf * 8 + ((j & 3) << 1);
}

__device__ __forceinline__ f32x8 wmma_bf16(const Frag& a, const Frag& b, f32x8 c) {
  return __builtin_amdgcn_wmma_f32_16x16x32_bf16(false, a.v, false, b.v, (short)0, c, false, false);
}

// -------------------- conversion kernels --------------------
__global__ void cvt_bf16(const float* __restrict__ in, unsigned short* __restrict__ out, int n) {
  int i = blockIdx.x * blockDim.x + threadIdx.x;
  if (i < n) out[i] = f32_to_bf16(in[i]);
}

// in [rows, cols] f32 -> out [cols, rows] bf16 (transpose)
__global__ void cvt_bf16_T(const float* __restrict__ in, unsigned short* __restrict__ out,
                           int rows, int cols) {
  int i = blockIdx.x * blockDim.x + threadIdx.x;
  if (i < rows * cols) {
    int c = i / rows;
    int r = i - c * rows;
    out[i] = f32_to_bf16(in[(size_t)r * cols + c]);
  }
}

// -------------------- QKV projection GEMM (64x64 tile / wave) --------------------
// xb [B*N, DIM] bf16, wT [3*H*DH, DIM] bf16 (W_qkv transposed)
// qo [B,H,N,DH] (pre-scaled by DH^-0.5), ko [B,H,N,DH], vTo [B,H,DH,N]
__global__ __launch_bounds__(32) void gemm_qkv_kernel(
    const unsigned short* __restrict__ xb,
    const unsigned short* __restrict__ wT,
    unsigned short* __restrict__ qo,
    unsigned short* __restrict__ ko,
    unsigned short* __restrict__ vTo)
{
  const int lane = threadIdx.x;
  const int mrow = lane & 15, hf = lane >> 4;
  const int n0 = blockIdx.x * 64;
  const int m0 = blockIdx.y * 64;

  f32x8 acc[4][4];
#pragma unroll
  for (int i = 0; i < 4; i++)
#pragma unroll
    for (int j = 0; j < 4; j++)
#pragma unroll
      for (int r = 0; r < 8; r++) acc[i][j][r] = 0.f;

  for (int kk = 0; kk < DIM_; kk += 32) {
    Frag af[4], bf[4];
#pragma unroll
    for (int i = 0; i < 4; i++) {
      const unsigned short* pa = xb + (size_t)(m0 + i * 16 + mrow) * DIM_ + kk;
      const unsigned short* pb = wT + (size_t)(n0 + i * 16 + mrow) * DIM_ + kk;
#pragma unroll
      for (int j = 0; j < 8; j++) {
        af[i].u[j] = *(const uint32_t*)(pa + koffA(j, hf));
        bf[i].u[j] = *(const uint32_t*)(pb + koffB(j, hf));
      }
    }
#pragma unroll
    for (int i = 0; i < 4; i++)
#pragma unroll
      for (int j = 0; j < 4; j++)
        acc[i][j] = wmma_bf16(af[i], bf[j], acc[i][j]);
  }

#pragma unroll
  for (int ti = 0; ti < 4; ti++)
#pragma unroll
    for (int tj = 0; tj < 4; tj++)
#pragma unroll
      for (int r = 0; r < 8; r++) {
        const int m = m0 + ti * 16 + r + 8 * hf;
        const int c = n0 + tj * 16 + mrow;
        const float val = acc[ti][tj][r];
        const int which = c >> 10;          // 0=q 1=k 2=v
        const int hh = (c & 1023) >> 6;
        const int d  = c & 63;
        const int bb = m >> 11;
        const int n  = m & 2047;
        const size_t head = (size_t)bb * H_ + hh;
        if (which == 0)
          qo[(head * N_ + n) * DH_ + d] = f32_to_bf16(val * 0.125f); // fold softmax scale
        else if (which == 1)
          ko[(head * N_ + n) * DH_ + d] = f32_to_bf16(val);
        else
          vTo[(head * DH_ + d) * N_ + n] = f32_to_bf16(val);
      }
}

// -------------------- flash attention --------------------
// one wave per (b, h, 32-query tile); K/V fragments reused across both 16-row
// sub-tiles; mask is L2-resident (64MB < 192MB L2)
__global__ __launch_bounds__(32) void flash_attn_kernel(
    const unsigned short* __restrict__ q,   // [B,H,N,DH]
    const unsigned short* __restrict__ k,   // [B,H,N,DH]
    const unsigned short* __restrict__ vT,  // [B,H,DH,N]
    const int* __restrict__ mask,           // [B,N,N]
    unsigned short* __restrict__ ctx)       // [B,N,H*DH]
{
  const int lane = threadIdx.x;
  const int mrow = lane & 15, hf = lane >> 4;
  const int q0 = blockIdx.x * 32;
  const int h  = blockIdx.y;
  const int b  = blockIdx.z;
  const size_t headoff = ((size_t)b * H_ + h) * (size_t)N_ * DH_;
  const unsigned short* qh = q  + headoff;
  const unsigned short* kh = k  + headoff;
  const unsigned short* vh = vT + headoff;
  const int* maskb = mask + (size_t)b * N_ * N_;

  // Q tiles -> A-frags: [qt][t] with d = t*32..t*32+31
  Frag aq[2][2];
#pragma unroll
  for (int qt = 0; qt < 2; qt++)
#pragma unroll
    for (int t = 0; t < 2; t++) {
      const unsigned short* pq = qh + (size_t)(q0 + qt * 16 + mrow) * DH_ + t * 32;
#pragma unroll
      for (int j = 0; j < 8; j++)
        aq[qt][t].u[j] = *(const uint32_t*)(pq + koffA(j, hf));
    }

  float m_i[2][8], l_i[2][8];
  f32x8 o[2][4];
#pragma unroll
  for (int qt = 0; qt < 2; qt++) {
#pragma unroll
    for (int r = 0; r < 8; r++) { m_i[qt][r] = -1e30f; l_i[qt][r] = 0.f; }
#pragma unroll
    for (int d = 0; d < 4; d++)
#pragma unroll
      for (int r = 0; r < 8; r++) o[qt][d][r] = 0.f;
  }

  __shared__ unsigned short pbuf[32][32];   // P tiles staging (C-layout -> A-layout)

  for (int kc = 0; kc < N_; kc += 32) {
    // prefetch next key/value chunk into caches (global_prefetch_b8)
    if (kc + 32 < N_) {
      __builtin_prefetch(kh + (size_t)(kc + 32 + mrow) * DH_, 0, 1);
      __builtin_prefetch(vh + (size_t)mrow * N_ + kc + 32, 0, 1);
    }

    // ---- S = Q @ K^T for 32 keys, both query sub-tiles ----
    f32x8 s[2][2];
#pragma unroll
    for (int qt = 0; qt < 2; qt++)
#pragma unroll
      for (int sub = 0; sub < 2; sub++)
#pragma unroll
        for (int r = 0; r < 8; r++) s[qt][sub][r] = 0.f;

#pragma unroll
    for (int sub = 0; sub < 2; sub++) {
      const int key0 = kc + sub * 16;
#pragma unroll
      for (int t = 0; t < 2; t++) {
        Frag bk;
        const unsigned short* pk = kh + (size_t)(key0 + mrow) * DH_ + t * 32;
#pragma unroll
        for (int j = 0; j < 8; j++)
          bk.u[j] = *(const uint32_t*)(pk + koffB(j, hf));
#pragma unroll
        for (int qt = 0; qt < 2; qt++)
          s[qt][sub] = wmma_bf16(aq[qt][t], bk, s[qt][sub]);
      }
    }

    // ---- mask + online softmax (rows r+8*hf of each sub-tile live in this lane) ----
#pragma unroll
    for (int qt = 0; qt < 2; qt++) {
#pragma unroll
      for (int r = 0; r < 8; r++) {
        const int qrow = q0 + qt * 16 + r + 8 * hf;
        const int* mp = maskb + (size_t)qrow * N_ + kc;
#pragma unroll
        for (int sub = 0; sub < 2; sub++)
          if (mp[sub * 16 + mrow] == 1) s[qt][sub][r] = -1e30f;

        float t = fmaxf(s[qt][0][r], s[qt][1][r]);
#pragma unroll
        for (int mm = 1; mm < 16; mm <<= 1)
          t = fmaxf(t, __shfl_xor(t, mm, 32));

        const float mn   = fmaxf(m_i[qt][r], t);
        const float corr = __expf(m_i[qt][r] - mn);
        float rs = 0.f;
#pragma unroll
        for (int sub = 0; sub < 2; sub++) {
          const float p = __expf(s[qt][sub][r] - mn);
          s[qt][sub][r] = p;
          rs += p;
        }
#pragma unroll
        for (int mm = 1; mm < 16; mm <<= 1)
          rs += __shfl_xor(rs, mm, 32);
        l_i[qt][r] = l_i[qt][r] * corr + rs;
        m_i[qt][r] = mn;
#pragma unroll
        for (int d = 0; d < 4; d++) o[qt][d][r] *= corr;
      }

      // stage P sub-tile to LDS (row-major bf16)
#pragma unroll
      for (int sub = 0; sub < 2; sub++)
#pragma unroll
        for (int r = 0; r < 8; r++)
          pbuf[qt * 16 + r + 8 * hf][sub * 16 + mrow] = f32_to_bf16(s[qt][sub][r]);
    }
    __syncthreads();   // single-wave WG: forces s_wait_dscnt before A-frag reads

    // re-read P as A-frags (C-layout -> A-layout transpose through LDS)
    Frag pa[2];
#pragma unroll
    for (int qt = 0; qt < 2; qt++)
#pragma unroll
      for (int j = 0; j < 8; j++)
        pa[qt].u[j] = *(const uint32_t*)(&pbuf[qt * 16 + mrow][0] + koffA(j, hf));

    // ---- O += P @ V (V transposed: contiguous along key); V frag reused 2x ----
#pragma unroll
    for (int dsub = 0; dsub < 4; dsub++) {
      Frag vf;
      const unsigned short* pv = vh + (size_t)(dsub * 16 + mrow) * N_ + kc;
#pragma unroll
      for (int j = 0; j < 8; j++)
        vf.u[j] = *(const uint32_t*)(pv + koffB(j, hf));
#pragma unroll
      for (int qt = 0; qt < 2; qt++)
        o[qt][dsub] = wmma_bf16(pa[qt], vf, o[qt][dsub]);
    }
  }

  // ---- normalize and write context [b, n, h*64+d] bf16 ----
#pragma unroll
  for (int qt = 0; qt < 2; qt++)
#pragma unroll
    for (int dsub = 0; dsub < 4; dsub++)
#pragma unroll
      for (int r = 0; r < 8; r++) {
        const int qrow = q0 + qt * 16 + r + 8 * hf;
        const float val = o[qt][dsub][r] / l_i[qt][r];
        ctx[((size_t)b * N_ + qrow) * (H_ * DH_) + h * DH_ + dsub * 16 + mrow] =
            f32_to_bf16(val);
      }
}

// -------------------- output projection GEMM (64x64 tile / wave) --------------------
// ctx [B*N, 1024] bf16, wT [1024, 1024] bf16 (= W_out^T), out f32 + bias
__global__ __launch_bounds__(32) void gemm_out_kernel(
    const unsigned short* __restrict__ ctx,
    const unsigned short* __restrict__ wT,
    const float* __restrict__ bias,
    float* __restrict__ out)
{
  const int lane = threadIdx.x;
  const int mrow = lane & 15, hf = lane >> 4;
  const int n0 = blockIdx.x * 64;
  const int m0 = blockIdx.y * 64;

  f32x8 acc[4][4];
#pragma unroll
  for (int i = 0; i < 4; i++)
#pragma unroll
    for (int j = 0; j < 4; j++)
#pragma unroll
      for (int r = 0; r < 8; r++) acc[i][j][r] = 0.f;

  for (int kk = 0; kk < DIM_; kk += 32) {
    Frag af[4], bf[4];
#pragma unroll
    for (int i = 0; i < 4; i++) {
      const unsigned short* pa = ctx + (size_t)(m0 + i * 16 + mrow) * DIM_ + kk;
      const unsigned short* pb = wT  + (size_t)(n0 + i * 16 + mrow) * DIM_ + kk;
#pragma unroll
      for (int j = 0; j < 8; j++) {
        af[i].u[j] = *(const uint32_t*)(pa + koffA(j, hf));
        bf[i].u[j] = *(const uint32_t*)(pb + koffB(j, hf));
      }
    }
#pragma unroll
    for (int i = 0; i < 4; i++)
#pragma unroll
      for (int j = 0; j < 4; j++)
        acc[i][j] = wmma_bf16(af[i], bf[j], acc[i][j]);
  }

#pragma unroll
  for (int ti = 0; ti < 4; ti++)
#pragma unroll
    for (int tj = 0; tj < 4; tj++)
#pragma unroll
      for (int r = 0; r < 8; r++) {
        const int m = m0 + ti * 16 + r + 8 * hf;
        const int c = n0 + tj * 16 + mrow;
        out[(size_t)m * DIM_ + c] = acc[ti][tj][r] + bias[c];
      }
}

// -------------------- host launcher --------------------
extern "C" void kernel_launch(void* const* d_in, const int* in_sizes, int n_in,
                              void* d_out, int out_size, void* d_ws, size_t ws_size,
                              hipStream_t stream) {
  const float* x    = (const float*)d_in[0];
  const int*   mask = (const int*)d_in[1];
  const float* Wqkv = (const float*)d_in[2];
  const float* Wout = (const float*)d_in[3];
  const float* bout = (const float*)d_in[4];
  float* out = (float*)d_out;
  (void)in_sizes; (void)n_in; (void)out_size; (void)ws_size;

  char* ws = (char*)d_ws;
  size_t off = 0;
  auto take = [&](size_t elems) {
    unsigned short* p = (unsigned short*)(ws + off);
    off = (off + elems * sizeof(unsigned short) + 255) & ~(size_t)255;
    return p;
  };
  const size_t MN = (size_t)B_ * N_;                 // 8192
  unsigned short* xb    = take(MN * DIM_);           // 16 MB
  unsigned short* wqkvT = take((size_t)3 * H_ * DH_ * DIM_);
  unsigned short* woutT = take((size_t)DIM_ * DIM_);
  unsigned short* qb    = take((size_t)B_ * H_ * N_ * DH_);
  unsigned short* kb    = take((size_t)B_ * H_ * N_ * DH_);
  unsigned short* vTb   = take((size_t)B_ * H_ * N_ * DH_);
  unsigned short* ctx   = take(MN * DIM_);

  {
    int n = (int)(MN * DIM_);
    cvt_bf16<<<(n + 255) / 256, 256, 0, stream>>>(x, xb, n);
  }
  {
    int n = DIM_ * 3 * H_ * DH_;
    cvt_bf16_T<<<(n + 255) / 256, 256, 0, stream>>>(Wqkv, wqkvT, DIM_, 3 * H_ * DH_);
  }
  {
    int n = DIM_ * DIM_;
    cvt_bf16_T<<<(n + 255) / 256, 256, 0, stream>>>(Wout, woutT, DIM_, DIM_);
  }

  gemm_qkv_kernel<<<dim3((3 * H_ * DH_) / 64, (unsigned)(MN / 64)), 32, 0, stream>>>(
      xb, wqkvT, qb, kb, vTb);

  flash_attn_kernel<<<dim3(N_ / 32, H_, B_), 32, 0, stream>>>(qb, kb, vTb, mask, ctx);

  gemm_out_kernel<<<dim3(DIM_ / 64, (unsigned)(MN / 64)), 32, 0, stream>>>(
      ctx, woutT, bout, out);
}
